// MultiHeadAttention_78245714199037
// MI455X (gfx1250) — compile-verified, matrix operand order checked by reading, not run on hardware
//
#include <hip/hip_runtime.h>

// ---------------------------------------------------------------------------
// CDNA5 (gfx1250) WMMA implementation of conv-projected multi-head attention.
// Wave32, V_WMMA_F32_16X16X32_F16 everywhere, full padded image staged in the
// 320KB LDS, B-matrix pre-swizzled into per-lane fragment order.
// Final conv stages its f16 input via GLOBAL_LOAD_ASYNC_TO_LDS (ASYNCcnt).
// ---------------------------------------------------------------------------

typedef _Float16 half16 __attribute__((ext_vector_type(16)));
typedef float    f32x8  __attribute__((ext_vector_type(8)));
typedef unsigned int u32x4 __attribute__((ext_vector_type(4)));
typedef unsigned int u32x2 __attribute__((ext_vector_type(2)));
typedef float    f32x4v __attribute__((ext_vector_type(4)));
typedef int      i32x4v __attribute__((ext_vector_type(4)));

union Frag16 { half16 v; u32x4 u[2]; };
union H8 { u32x4 u; _Float16 h[8]; };
union H4 { u32x2 u; _Float16 h[4]; };

#define WMMA_F16(A, B, C) \
  __builtin_amdgcn_wmma_f32_16x16x32_f16(false, (A), false, (B), (short)0, (C), false, false)

// gfx1250 async global->LDS copy path (ASYNCcnt), guarded so the file stays
// compile-clean if this toolchain doesn't declare the builtins.
#if __has_builtin(__builtin_amdgcn_global_load_async_to_lds_b128) && \
    __has_builtin(__builtin_amdgcn_s_wait_asynccnt)
#define HAVE_ASYNC_LDS 1
#else
#define HAVE_ASYNC_LDS 0
#endif

// Builtin prototype (from hipcc diagnostic): (int4 as1*, int4 as3*, Imm, Imm)
#define AS_G_B128(p) ((__attribute__((address_space(1))) i32x4v*)(p))
#define AS_L_B128(p) ((__attribute__((address_space(3))) i32x4v*)(p))

// Shapes
#define BATCH 8
#define SEQ 32
#define NIMG 256           // BATCH*SEQ
#define DM 256             // d_model
#define HEADS 8
#define DEPTH 32
#define RWD 8192           // 16*16*32 per-head flattened spatial*depth
#define KTOT 2304          // 3*3*256
#define KCHUNKS 72         // KTOT/32
#define PADW 18            // 16 + 2 halo
#define SMEM_CONV (PADW*PADW*DM*2)   // 165,888 bytes of f16 padded image

// ---------------------------------------------------------------------------
// Rearrange one conv weight (3,3,256,256) f32 HWIO into f16 B-fragment order:
// wf[((kc*16 + ntile)*32 + lane)*16 + j]  where per-lane K indices follow the
// 16-bit 32x16 B layout: base = lane<16 ? 0 : 8;
//   j<8  -> k = kc*32 + base + j
//   j>=8 -> k = kc*32 + 16 + base + (j-8)
// and N = ntile*16 + (lane&15).
// ---------------------------------------------------------------------------
__global__ __launch_bounds__(256)
void prep_weights_k(const float* __restrict__ w, _Float16* __restrict__ wf) {
  const int total = KCHUNKS * 16 * 32 * 16;   // 589,824
  for (int i = blockIdx.x * blockDim.x + threadIdx.x; i < total;
       i += gridDim.x * blockDim.x) {
    int j    = i & 15;
    int lane = (i >> 4) & 31;
    int nt   = (i >> 9) & 15;
    int kc   = i >> 13;
    int base = (lane < 16) ? 0 : 8;
    int k    = kc * 32 + ((j < 8) ? (base + j) : (16 + base + (j - 8)));
    int n    = nt * 16 + (lane & 15);
    wf[i] = (_Float16)w[(size_t)k * DM + n];
  }
}

// ---------------------------------------------------------------------------
// Implicit-GEMM 3x3 conv: one workgroup per image (16x16x256).
// 512 threads = 16 waves in a 4x4 grid; each wave owns a 64x64 tile = 4x4 WMMAs.
// Entire zero-padded image (18x18x256 f16 = 162KB) is staged in LDS.
// ---------------------------------------------------------------------------
template <bool IN_F16, bool RELU, bool OUT_ATTN>
__global__ __launch_bounds__(512)
void conv_gemm_k(const void* __restrict__ xin,
                 const _Float16* __restrict__ wf,
                 const float* __restrict__ bias,
                 _Float16* __restrict__ out_h,
                 float* __restrict__ out_f) {
  extern __shared__ __align__(16) char dynsmem[];
  _Float16* xs = (_Float16*)dynsmem;          // [18][18][256]

  const int tid = threadIdx.x;
  const int img = blockIdx.x;                 // b = img/32, s = img%32

  // Zero the padded image (covers the halo), then fill the interior.
  u32x4 z4 = {0u, 0u, 0u, 0u};
  for (int i = tid; i < SMEM_CONV / 16; i += 512) ((u32x4*)xs)[i] = z4;
  __syncthreads();

  if (IN_F16) {
    const _Float16* xh = (const _Float16*)xin + (size_t)img * 65536;
    for (int it = tid; it < 8192; it += 512) {
      int e0 = it << 3;
      int r = e0 >> 12, rem = e0 & 4095, c = rem >> 8, ci = rem & 255;
      _Float16* dst = xs + (((r + 1) * PADW + (c + 1)) << 8) + ci;
#if HAVE_ASYNC_LDS
      // gfx1250: async DMA-style copy, LDS write tracked by ASYNCcnt
      __builtin_amdgcn_global_load_async_to_lds_b128(
          AS_G_B128(xh + e0), AS_L_B128(dst), 0, 0);
#else
      *(u32x4*)dst = *(const u32x4*)(xh + e0);
#endif
    }
#if HAVE_ASYNC_LDS
    __builtin_amdgcn_s_wait_asynccnt(0);
#endif
  } else {
    const float* xf = (const float*)xin + (size_t)img * 65536;
    for (int it = tid; it < 16384; it += 512) {
      int e0 = it << 2;
      int r = e0 >> 12, rem = e0 & 4095, c = rem >> 8, ci = rem & 255;
      f32x4v f = *(const f32x4v*)(xf + e0);
      H4 h;
      h.h[0] = (_Float16)f.x; h.h[1] = (_Float16)f.y;
      h.h[2] = (_Float16)f.z; h.h[3] = (_Float16)f.w;
      *(u32x2*)(xs + (((r + 1) * PADW + (c + 1)) << 8) + ci) = h.u;
    }
  }
  __syncthreads();

  const int lane = tid & 31;
  const int w    = tid >> 5;       // 0..15
  const int wm   = w >> 2, wn = w & 3;
  const int lr   = lane & 15;
  const int base = (lane < 16) ? 0 : 8;
  const int mhi  = (lane < 16) ? 0 : 8;

  f32x8 zero = {0.f, 0.f, 0.f, 0.f, 0.f, 0.f, 0.f, 0.f};
  f32x8 acc[4][4];
#pragma unroll
  for (int i = 0; i < 4; ++i)
#pragma unroll
    for (int j = 0; j < 4; ++j) acc[i][j] = zero;

  for (int kc = 0; kc < KCHUNKS; ++kc) {
    int kyx  = kc >> 3;
    int ky   = kyx / 3, kx = kyx - ky * 3;
    int coff = (kc & 7) << 5;

    Frag16 af[4];
#pragma unroll
    for (int i = 0; i < 4; ++i) {
      int m    = (wm << 6) + (i << 4) + lr;
      int addr = (((m >> 4) + ky) * PADW + (m & 15) + kx) * DM + coff + base;
      af[i].u[0] = *(const u32x4*)(xs + addr);
      af[i].u[1] = *(const u32x4*)(xs + addr + 16);
    }
    Frag16 bf_[4];
#pragma unroll
    for (int j = 0; j < 4; ++j) {
      size_t off = ((((size_t)kc * 16) + (wn << 2) + j) * 32 + lane) << 4;
      bf_[j].u[0] = *(const u32x4*)(wf + off);
      bf_[j].u[1] = *(const u32x4*)(wf + off + 8);
    }
    if (kc + 1 < KCHUNKS)   // gfx1250 global_prefetch_b8 for next weight chunk
      __builtin_prefetch(wf + (((((size_t)(kc + 1) * 16) + (wn << 2)) * 32 + lane) << 4), 0, 0);

#pragma unroll
    for (int i = 0; i < 4; ++i)
#pragma unroll
      for (int j = 0; j < 4; ++j)
        acc[i][j] = WMMA_F16(af[i].v, bf_[j].v, acc[i][j]);
  }

  const int bimg = img >> 5, simg = img & 31;
#pragma unroll
  for (int i = 0; i < 4; ++i) {
#pragma unroll
    for (int j = 0; j < 4; ++j) {
      int n = (wn << 6) + (j << 4) + lr;
      float bv = bias[n];
#pragma unroll
      for (int v = 0; v < 8; ++v) {
        int m = (wm << 6) + (i << 4) + mhi + v;
        float val = acc[i][j][v] + bv;
        if (RELU) val = fmaxf(val, 0.f);
        if (OUT_ATTN) {
          // f16, head-split layout (b, h, s, rwd) for the attention GEMMs
          int head = n >> 5, d = n & 31;
          size_t idx = ((size_t)((bimg << 3) + head) * SEQ + simg) * RWD +
                       (size_t)m * DEPTH + d;
          out_h[idx] = (_Float16)val;
        } else {
          // f32, plain (b, s, r, c, cout) layout = final output
          out_f[(size_t)img * 65536 + (size_t)m * DM + n] = val;
        }
      }
    }
  }
}

// ---------------------------------------------------------------------------
// Attention: one block per (b,h). 256 threads = 8 waves.
// Phase 1: scores(32x32) = Q K^T over K=8192, split across waves, LDS reduce,
//          softmax by wave 0 (f32 probs -> d_out tail, f16 probs -> LDS).
// Phase 2: out = P V over N=8192, V staged transposed in LDS per 256-col chunk.
// ---------------------------------------------------------------------------
__global__ __launch_bounds__(256)
void attention_k(const _Float16* __restrict__ qp,
                 const _Float16* __restrict__ kp,
                 const _Float16* __restrict__ vp,
                 float* __restrict__ attn_out,
                 _Float16* __restrict__ oat) {
  __shared__ __align__(16) float     sred[8][2][2][8][32]; // 32KB partial scores
  __shared__ __align__(16) float     S[32][32];            // 4KB scores
  __shared__ __align__(16) _Float16  P[32][32];            // 2KB probs (f16)
  __shared__ __align__(16) _Float16  vT[256][32];          // 16KB V^T chunk

  const int tid  = threadIdx.x;
  const int lane = tid & 31;
  const int w    = tid >> 5;
  const int lr   = lane & 15;
  const int base = (lane < 16) ? 0 : 8;
  const int mhi  = (lane < 16) ? 0 : 8;
  const int bh   = blockIdx.x;              // b*8 + h

  const _Float16* qb = qp + (size_t)bh * (SEQ * RWD);
  const _Float16* kb = kp + (size_t)bh * (SEQ * RWD);
  const _Float16* vb = vp + (size_t)bh * (SEQ * RWD);

  f32x8 zero = {0.f, 0.f, 0.f, 0.f, 0.f, 0.f, 0.f, 0.f};
  f32x8 accS[2][2];
#pragma unroll
  for (int mi = 0; mi < 2; ++mi)
#pragma unroll
    for (int nj = 0; nj < 2; ++nj) accS[mi][nj] = zero;

  // ---- Phase 1: Q K^T, K chunks strided across the 8 waves ----
  for (int kc = w; kc < 256; kc += 8) {
    Frag16 aQ[2], bK[2];
#pragma unroll
    for (int t = 0; t < 2; ++t) {
      const _Float16* pq = qb + (size_t)((t << 4) + lr) * RWD + (kc << 5) + base;
      aQ[t].u[0] = *(const u32x4*)pq;
      aQ[t].u[1] = *(const u32x4*)(pq + 16);
      const _Float16* pk = kb + (size_t)((t << 4) + lr) * RWD + (kc << 5) + base;
      bK[t].u[0] = *(const u32x4*)pk;
      bK[t].u[1] = *(const u32x4*)(pk + 16);
    }
#pragma unroll
    for (int mi = 0; mi < 2; ++mi)
#pragma unroll
      for (int nj = 0; nj < 2; ++nj)
        accS[mi][nj] = WMMA_F16(aQ[mi].v, bK[nj].v, accS[mi][nj]);
  }
#pragma unroll
  for (int mi = 0; mi < 2; ++mi)
#pragma unroll
    for (int nj = 0; nj < 2; ++nj)
#pragma unroll
      for (int v = 0; v < 8; ++v) sred[w][mi][nj][v][lane] = accS[mi][nj][v];
  __syncthreads();

  // Cross-wave reduce into S[m][n] with the C-tile layout inverted.
  for (int e = tid; e < 1024; e += 256) {
    int m = e >> 5, n = e & 31;
    int mi = m >> 4, mm = m & 15, v = mm & 7, hi = mm >> 3;
    int nj = n >> 4, ln = (n & 15) + (hi << 4);
    float s = 0.f;
#pragma unroll
    for (int ww = 0; ww < 8; ++ww) s += sred[ww][mi][nj][v][ln];
    S[m][n] = s * 0.17677669529663687f;   // 1/sqrt(32)
  }
  __syncthreads();

  // Softmax, one row per lane of wave 0.
  if (tid < 32) {
    int m = tid;
    float mx = -1e30f;
    for (int n = 0; n < 32; ++n) mx = fmaxf(mx, S[m][n]);
    float sum = 0.f;
    for (int n = 0; n < 32; ++n) {
      float e = __expf(S[m][n] - mx);
      S[m][n] = e;
      sum += e;
    }
    float inv = 1.f / sum;
    for (int n = 0; n < 32; ++n) {
      float p = S[m][n] * inv;
      P[m][n] = (_Float16)p;
      attn_out[(size_t)bh * 1024 + m * 32 + n] = p;
    }
  }
  __syncthreads();

  // A fragments of P (single K chunk of 32), reused for all N tiles.
  Frag16 aP[2];
#pragma unroll
  for (int t = 0; t < 2; ++t) {
    const _Float16* pp = &P[(t << 4) + lr][base];
    aP[t].u[0] = *(const u32x4*)pp;
    aP[t].u[1] = *(const u32x4*)(pp + 16);
  }

  // ---- Phase 2: out = P V, N=8192 in 32 chunks of 256 columns ----
  const int bat = bh >> 3, hd = bh & 7;
  for (int st = 0; st < 32; ++st) {
    int x0 = st << 8;
    __syncthreads();               // previous chunk's readers are done
    for (int it = tid; it < 1024; it += 256) {
      int s = it >> 5, u = it & 31;
      H8 t8;
      t8.u = *(const u32x4*)(vb + (size_t)s * RWD + x0 + (u << 3));
#pragma unroll
      for (int e = 0; e < 8; ++e) vT[(u << 3) + e][s] = t8.h[e];
    }
    __syncthreads();

#pragma unroll
    for (int jt = 0; jt < 2; ++jt) {
      int nt = (w << 1) + jt;      // 16 n-tiles over 8 waves
      const _Float16* bp = &vT[(nt << 4) + lr][base];
      Frag16 bF;
      bF.u[0] = *(const u32x4*)bp;
      bF.u[1] = *(const u32x4*)(bp + 16);
#pragma unroll
      for (int mi = 0; mi < 2; ++mi) {
        f32x8 o = WMMA_F16(aP[mi].v, bF.v, zero);
#pragma unroll
        for (int v = 0; v < 8; ++v) {
          int sq = (mi << 4) + mhi + v;
          int xg = x0 + (nt << 4) + lr;
          int rc = xg >> 5, d = xg & 31;
          // f16 plain (b, s, r, c, cout) layout for the final conv
          oat[((size_t)(bat * SEQ + sq) * 256 + rc) * DM + hd * DEPTH + d] =
              (_Float16)o[v];
        }
      }
    }
  }
}

// ---------------------------------------------------------------------------
// Host-side orchestration (graph-capture safe: only kernel launches).
// ---------------------------------------------------------------------------
extern "C" void kernel_launch(void* const* d_in, const int* in_sizes, int n_in,
                              void* d_out, int out_size, void* d_ws, size_t ws_size,
                              hipStream_t stream) {
  (void)in_sizes; (void)n_in; (void)out_size; (void)ws_size;

  const float* q  = (const float*)d_in[0];
  const float* k  = (const float*)d_in[1];
  const float* v  = (const float*)d_in[2];
  const float* wq = (const float*)d_in[3];
  const float* bq = (const float*)d_in[4];
  const float* wk = (const float*)d_in[5];
  const float* bk = (const float*)d_in[6];
  const float* wv = (const float*)d_in[7];
  const float* bv = (const float*)d_in[8];
  const float* wf = (const float*)d_in[9];
  const float* bf = (const float*)d_in[10];

  char* ws = (char*)d_ws;
  const size_t WFRAG = (size_t)KTOT * DM * 2;        // 1,179,648 B each
  const size_t TENS  = (size_t)NIMG * 65536 * 2;     // 33,554,432 B each (f16)
  _Float16* wfragQ = (_Float16*)(ws + 0 * WFRAG);
  _Float16* wfragK = (_Float16*)(ws + 1 * WFRAG);
  _Float16* wfragV = (_Float16*)(ws + 2 * WFRAG);
  _Float16* wfragF = (_Float16*)(ws + 3 * WFRAG);
  size_t off = 4 * WFRAG;
  _Float16* qp  = (_Float16*)(ws + off); off += TENS;
  _Float16* kp  = (_Float16*)(ws + off); off += TENS;
  _Float16* vp  = (_Float16*)(ws + off); off += TENS;
  _Float16* oat = (_Float16*)(ws + off); off += TENS;

  float* outc = (float*)d_out;
  float* attn = outc + (size_t)NIMG * 65536;   // 16,777,216 floats, then attn

  // 1) weights -> f16 B-fragment order
  prep_weights_k<<<dim3(1152), dim3(256), 0, stream>>>(wq, wfragQ);
  prep_weights_k<<<dim3(1152), dim3(256), 0, stream>>>(wk, wfragK);
  prep_weights_k<<<dim3(1152), dim3(256), 0, stream>>>(wv, wfragV);
  prep_weights_k<<<dim3(1152), dim3(256), 0, stream>>>(wf, wfragF);

  // 2) q/k/v projections: conv + bias + relu, f16 head-split output
  conv_gemm_k<false, true, true><<<dim3(NIMG), dim3(512), SMEM_CONV, stream>>>(
      q, wfragQ, bq, qp, nullptr);
  conv_gemm_k<false, true, true><<<dim3(NIMG), dim3(512), SMEM_CONV, stream>>>(
      k, wfragK, bk, kp, nullptr);
  conv_gemm_k<false, true, true><<<dim3(NIMG), dim3(512), SMEM_CONV, stream>>>(
      v, wfragV, bv, vp, nullptr);

  // 3) attention per (b,h): scores, softmax (f32 attn out), P·V (f16 out)
  attention_k<<<dim3(64), dim3(256), 0, stream>>>(qp, kp, vp, attn, oat);

  // 4) final conv: f16 in, no relu, f32 plain output
  conv_gemm_k<true, false, false><<<dim3(NIMG), dim3(512), SMEM_CONV, stream>>>(
      oat, wfragF, bf, nullptr, outc);
}
